// TemporalGNN_22952305229948
// MI455X (gfx1250) — matrix operand order repaired and checked by reading, not compiled, not run on hardware
//
#include <hip/hip_runtime.h>
#include <hip/hip_bf16.h>

// TemporalGNN for MI455X (gfx1250): DirGCN x2 -> LSTM -> predictor.
// v3: GEMM templatized on DUAL so the WMMA inner loop is branch-free
// straight-line code -> the scheduler can clause the b128 loads and overlap
// them with WMMAs via staged loadcnt waits. Unconditional speculative
// prefetch of the next A K-panel (global_prefetch_b8).

constexpr int TS = 16;       // timesteps
constexpr int NN = 10000;    // nodes (= 125 * 80 = 625 * 16)
constexpr int FF = 64;       // input features
constexpr int HH = 256;      // hidden
constexpr int EE = 320000;   // edges per step (= 1250 * 256)
constexpr int MT = 5;        // M-tiles per wave (80 rows); 625 % 5 == 0

typedef __attribute__((ext_vector_type(16))) __bf16 v16bf;
typedef __attribute__((ext_vector_type(8)))  __bf16 v8bf;
typedef __attribute__((ext_vector_type(4)))  __bf16 v4bf;
typedef __attribute__((ext_vector_type(8)))  float  v8f;

__device__ inline __bf16 to_bf16(float f) {
  union { float f; unsigned u; } v; v.f = f;
  unsigned r = v.u + 0x7FFFu + ((v.u >> 16) & 1u);   // round-to-nearest-even
  unsigned short h = (unsigned short)(r >> 16);
  return __builtin_bit_cast(__bf16, h);
}

__device__ inline v8f wmma_bf16(v16bf a, v16bf b, v8f c) {
  return __builtin_amdgcn_wmma_f32_16x16x32_bf16(
      /*neg_a=*/false, a, /*neg_b=*/false, b,
      /*c_mod=*/(short)0, c, /*reuse_a=*/false, /*reuse_b=*/false);
}

// ---------------------------------------------------- f32 -> bf16 convert ----
// Vectorized: 4 floats per thread. n4 = n/4 (all sizes here divisible by 4).
__global__ void cvt_f32_bf16_k(const float* __restrict__ src,
                               __bf16* __restrict__ dst, int n4, float scale) {
  int i = blockIdx.x * 256 + threadIdx.x;
  if (i < n4) {
    float4 v = ((const float4*)src)[i];
    v4bf o;
    o[0] = to_bf16(v.x * scale);
    o[1] = to_bf16(v.y * scale);
    o[2] = to_bf16(v.z * scale);
    o[3] = to_bf16(v.w * scale);
    ((v4bf*)dst)[i] = o;
  }
}

__global__ void combine_bias_k(const float* __restrict__ a,
                               const float* __restrict__ b,
                               float* __restrict__ o, int n, float s) {
  int i = blockIdx.x * 256 + threadIdx.x;
  if (i < n) o[i] = s * (a[i] + b[i]);
}

// ---------------------------------------------------------------- degrees ----
__global__ void deg_k(const int* __restrict__ src, const int* __restrict__ dst,
                      float* __restrict__ dout, float* __restrict__ din) {
  int e = blockIdx.x * 256 + threadIdx.x;             // EE exact multiple
  unsafeAtomicAdd(&dout[src[e]], 1.0f);               // global_atomic_add_f32
  unsafeAtomicAdd(&din[dst[e]], 1.0f);
}

__global__ void inv_sqrt_k(const float* __restrict__ deg,
                           float* __restrict__ o) {
  int i = blockIdx.x * 256 + threadIdx.x;
  if (i < NN) { float d = deg[i]; o[i] = d > 0.0f ? rsqrtf(d) : 0.0f; }
}

// ---------------------------------------------------------------- scatter ----
// grid: (EE/256, dim/4). Each thread handles one edge x 4 features, both
// directions: agg[src] += nrm*x[dst], aggT[dst] += nrm*x[src].
__global__ void scatter_k(const float* __restrict__ x,
                          const int* __restrict__ src,
                          const int* __restrict__ dst,
                          const float* __restrict__ oi,
                          const float* __restrict__ ii,
                          float* __restrict__ agg, float* __restrict__ aggT,
                          int dim) {
  int e = blockIdx.x * 256 + threadIdx.x;             // < EE exact
  int c = blockIdx.y * 4;
  int s = src[e], d = dst[e];
  float nrm = oi[s] * ii[d];
  const float4 xd = *(const float4*)(x + (size_t)d * dim + c);
  const float4 xs = *(const float4*)(x + (size_t)s * dim + c);
  float* pa = agg  + (size_t)s * dim + c;
  float* pt = aggT + (size_t)d * dim + c;
  unsafeAtomicAdd(pa + 0, nrm * xd.x);
  unsafeAtomicAdd(pa + 1, nrm * xd.y);
  unsafeAtomicAdd(pa + 2, nrm * xd.z);
  unsafeAtomicAdd(pa + 3, nrm * xd.w);
  unsafeAtomicAdd(pt + 0, nrm * xs.x);
  unsafeAtomicAdd(pt + 1, nrm * xs.y);
  unsafeAtomicAdd(pt + 2, nrm * xs.z);
  unsafeAtomicAdd(pt + 3, nrm * xs.w);
}

// ------------------------------------------------------------------- GEMM ----
// OUT[N,Hout] = epi( A1[N,K] @ W1[Hout,K]^T [+ A2 @ W2^T] + bias ), bf16 in,
// f32 accumulate. One wave per (80-row M-strip x 16-col H-tile): B fragments
// loaded once per K-step and reused by MT=5 A fragments. DUAL is a template
// parameter so the K-loop body is branch-free straight-line code (loads can
// clause and overlap WMMAs via staged loadcnt waits).
// A layout (16-bit 16x32): lanes 0-15 row M hold K 0-15, lanes 16-31
// hold K 16-31 (matches B) -> two contiguous v8bf (16B) loads per fragment.
// B layout: lane = output column, lanes 0-15 K 0-15, lanes 16-31 K 16-31
// -> one contiguous v16bf (32B) load per lane.
__device__ inline v16bf load_a_frag(const __bf16* __restrict__ base) {
  v8bf lo = *(const v8bf*)(base);
  v8bf hi = *(const v8bf*)(base + 8);
  return __builtin_shufflevector(lo, hi, 0, 1, 2, 3, 4, 5, 6, 7,
                                 8, 9, 10, 11, 12, 13, 14, 15);
}

template <bool DUAL>
__global__ void __launch_bounds__(32)
gemm_k(const __bf16* __restrict__ A1, const __bf16* __restrict__ A2,
       const __bf16* __restrict__ W1, const __bf16* __restrict__ W2,
       const float* __restrict__ bias, float* __restrict__ out,
       int K, int Hout, int relu) {
  const int lane  = threadIdx.x;
  const int l15   = lane & 15;
  const int khalf = lane >> 4;
  const int hcol  = blockIdx.x * 16 + l15;       // output column / W row
  const int bko   = khalf * 16;                  // B k-offset for this lane
  const int rowb  = blockIdx.y * (16 * MT);      // first row of M-strip

  v8f acc[MT];
#pragma unroll
  for (int m = 0; m < MT; ++m) acc[m] = (v8f){};

  const size_t wrow = (size_t)hcol * K;
  for (int kb = 0; kb < K; kb += 32) {
    const v16bf b1 = *(const v16bf*)(W1 + wrow + kb + bko);
    v16bf b2;
    if (DUAL) b2 = *(const v16bf*)(W2 + wrow + kb + bko);
#pragma unroll
    for (int m = 0; m < MT; ++m) {
      const size_t abase = (size_t)(rowb + m * 16 + l15) * K + kb + khalf * 16;
      v16bf a = load_a_frag(A1 + abase);
      __builtin_prefetch(A1 + abase + 32, 0, 1);  // speculative: safe past end
      acc[m] = wmma_bf16(a, b1, acc[m]);
      if (DUAL) {
        v16bf a2 = load_a_frag(A2 + abase);
        acc[m] = wmma_bf16(a2, b2, acc[m]);
      }
    }
  }
  // C/D layout: VGPR r, lanes 0-15 -> M=r, lanes 16-31 -> M=r+8; N = lane&15.
  const float bv = bias[hcol];
#pragma unroll
  for (int m = 0; m < MT; ++m) {
#pragma unroll
    for (int r = 0; r < 8; ++r) {
      int row = rowb + m * 16 + r + khalf * 8;
      float v = acc[m][r] + bv;
      if (relu) v = fmaxf(v, 0.0f);
      out[(size_t)row * Hout + blockIdx.x * 16 + l15] = v;
    }
  }
}

// ------------------------------------------------------------------- LSTM ----
__global__ void lstm_update_k(const float* __restrict__ gates,
                              float* __restrict__ h, float* __restrict__ c) {
  int idx = blockIdx.x * 256 + threadIdx.x;  // NN*HH exact
  int n = idx >> 8;                           // HH == 256
  int j = idx & 255;
  const float* g = gates + (size_t)n * 1024;
  float ig = g[j], fg = g[256 + j], gg = g[512 + j], og = g[768 + j];
  float si = 1.0f / (1.0f + __expf(-ig));
  float sf = 1.0f / (1.0f + __expf(-fg));
  float so = 1.0f / (1.0f + __expf(-og));
  float cv = sf * c[idx] + si * tanhf(gg);
  c[idx] = cv;
  h[idx] = so * tanhf(cv);
}

// ----------------------------------------------------------------- launch ----
extern "C" void kernel_launch(void* const* d_in, const int* in_sizes, int n_in,
                              void* d_out, int out_size, void* d_ws, size_t ws_size,
                              hipStream_t stream) {
  const float* x_seq = (const float*)d_in[0];
  const int*   eidx  = (const int*)d_in[1];
  const float* W1s = (const float*)d_in[2];  const float* b1s = (const float*)d_in[3];
  const float* W1d = (const float*)d_in[4];  const float* b1d = (const float*)d_in[5];
  const float* W2s = (const float*)d_in[6];  const float* b2s = (const float*)d_in[7];
  const float* W2d = (const float*)d_in[8];  const float* b2d = (const float*)d_in[9];
  const float* Wih = (const float*)d_in[10]; const float* Whh = (const float*)d_in[11];
  const float* bih = (const float*)d_in[12]; const float* bhh = (const float*)d_in[13];
  const float* Wp  = (const float*)d_in[14]; const float* bp  = (const float*)d_in[15];

  // ---- carve workspace (256B aligned chunks) ----
  char* ws = (char*)d_ws;
  auto carve = [&](size_t bytes) { char* p = ws; ws += (bytes + 255) & ~(size_t)255; return p; };
  float* deg_out = (float*)carve(NN * 4);
  float* deg_in  = (float*)carve(NN * 4);
  float* oi      = (float*)carve(NN * 4);
  float* ii      = (float*)carve(NN * 4);
  float* agg     = (float*)carve((size_t)NN * HH * 4);   // f32 scatter targets
  float* aggT    = (float*)carve((size_t)NN * HH * 4);
  float* h1      = (float*)carve((size_t)NN * HH * 4);   // layer-1 output (f32)
  float* gbuf    = (float*)carve((size_t)NN * HH * 4);   // layer-2 output (f32)
  float* gates   = (float*)carve((size_t)NN * 4 * HH * 4);
  float* hstate  = (float*)carve((size_t)NN * HH * 4);
  float* cstate  = (float*)carve((size_t)NN * HH * 4);
  __bf16* aggb    = (__bf16*)carve((size_t)NN * HH * 2); // bf16 GEMM inputs
  __bf16* aggTb   = (__bf16*)carve((size_t)NN * HH * 2);
  __bf16* gbufb   = (__bf16*)carve((size_t)NN * HH * 2);
  __bf16* hstateb = (__bf16*)carve((size_t)NN * HH * 2);
  __bf16* Wc1s = (__bf16*)carve((size_t)HH * FF * 2);
  __bf16* Wc1d = (__bf16*)carve((size_t)HH * FF * 2);
  __bf16* Wc2s = (__bf16*)carve((size_t)HH * HH * 2);
  __bf16* Wc2d = (__bf16*)carve((size_t)HH * HH * 2);
  __bf16* Wihb = (__bf16*)carve((size_t)4 * HH * HH * 2);
  __bf16* Whhb = (__bf16*)carve((size_t)4 * HH * HH * 2);
  __bf16* Wpb  = (__bf16*)carve((size_t)FF * HH * 2);
  float* bc1   = (float*)carve(HH * 4);
  float* bc2   = (float*)carve(HH * 4);
  float* blstm = (float*)carve(4 * HH * 4);

  auto blk4 = [](int n) { return (n / 4 + 255) / 256; };       // for cvt (n%4==0)
  auto blk  = [](int n) { return (n + 255) / 256; };

  // ---- weight prep (alpha=0.5 folded into GCN weights/biases) ----
  cvt_f32_bf16_k<<<blk4(HH * FF), 256, 0, stream>>>(W1s, Wc1s, HH * FF / 4, 0.5f);
  cvt_f32_bf16_k<<<blk4(HH * FF), 256, 0, stream>>>(W1d, Wc1d, HH * FF / 4, 0.5f);
  cvt_f32_bf16_k<<<blk4(HH * HH), 256, 0, stream>>>(W2s, Wc2s, HH * HH / 4, 0.5f);
  cvt_f32_bf16_k<<<blk4(HH * HH), 256, 0, stream>>>(W2d, Wc2d, HH * HH / 4, 0.5f);
  cvt_f32_bf16_k<<<blk4(4 * HH * HH), 256, 0, stream>>>(Wih, Wihb, 4 * HH * HH / 4, 1.0f);
  cvt_f32_bf16_k<<<blk4(4 * HH * HH), 256, 0, stream>>>(Whh, Whhb, 4 * HH * HH / 4, 1.0f);
  cvt_f32_bf16_k<<<blk4(FF * HH), 256, 0, stream>>>(Wp, Wpb, FF * HH / 4, 1.0f);
  combine_bias_k<<<blk(HH), 256, 0, stream>>>(b1s, b1d, bc1, HH, 0.5f);
  combine_bias_k<<<blk(HH), 256, 0, stream>>>(b2s, b2d, bc2, HH, 0.5f);
  combine_bias_k<<<blk(4 * HH), 256, 0, stream>>>(bih, bhh, blstm, 4 * HH, 1.0f);

  hipMemsetAsync(hstateb, 0, (size_t)NN * HH * 2, stream);  // bf16 zero == 0x0000
  hipMemsetAsync(cstate, 0, (size_t)NN * HH * 4, stream);

  const dim3 gemm_grid_H(HH / 16, NN / (16 * MT));      // (16, 125)
  const dim3 gemm_grid_G(4 * HH / 16, NN / (16 * MT));  // (64, 125)
  const dim3 gemm_grid_P(FF / 16, NN / (16 * MT));      // (4, 125)

  // ---- fused timestep loop: GCN1 -> GCN2 -> LSTM step ----
  for (int t = 0; t < TS; ++t) {
    const int* src = eidx + (size_t)t * 2 * EE;
    const int* dst = src + EE;
    const float* xt = x_seq + (size_t)t * NN * FF;

    hipMemsetAsync(deg_out, 0, NN * 4, stream);
    hipMemsetAsync(deg_in, 0, NN * 4, stream);
    deg_k<<<EE / 256, 256, 0, stream>>>(src, dst, deg_out, deg_in);
    inv_sqrt_k<<<blk(NN), 256, 0, stream>>>(deg_out, oi);
    inv_sqrt_k<<<blk(NN), 256, 0, stream>>>(deg_in, ii);

    // layer 1: [N,F] aggregate -> bf16 -> dual GEMM (F -> H) + ReLU
    hipMemsetAsync(agg, 0, (size_t)NN * FF * 4, stream);
    hipMemsetAsync(aggT, 0, (size_t)NN * FF * 4, stream);
    scatter_k<<<dim3(EE / 256, FF / 4), 256, 0, stream>>>(xt, src, dst, oi, ii, agg, aggT, FF);
    cvt_f32_bf16_k<<<blk4(NN * FF), 256, 0, stream>>>(agg, aggb, NN * FF / 4, 1.0f);
    cvt_f32_bf16_k<<<blk4(NN * FF), 256, 0, stream>>>(aggT, aggTb, NN * FF / 4, 1.0f);
    gemm_k<true><<<gemm_grid_H, 32, 0, stream>>>(aggb, aggTb, Wc1s, Wc1d, bc1, h1, FF, HH, 1);

    // layer 2: [N,H] aggregate -> bf16 -> dual GEMM (H -> H)
    hipMemsetAsync(agg, 0, (size_t)NN * HH * 4, stream);
    hipMemsetAsync(aggT, 0, (size_t)NN * HH * 4, stream);
    scatter_k<<<dim3(EE / 256, HH / 4), 256, 0, stream>>>(h1, src, dst, oi, ii, agg, aggT, HH);
    cvt_f32_bf16_k<<<blk4(NN * HH), 256, 0, stream>>>(agg, aggb, NN * HH / 4, 1.0f);
    cvt_f32_bf16_k<<<blk4(NN * HH), 256, 0, stream>>>(aggT, aggTb, NN * HH / 4, 1.0f);
    gemm_k<true><<<gemm_grid_H, 32, 0, stream>>>(aggb, aggTb, Wc2s, Wc2d, bc2, gbuf, HH, HH, 0);

    // LSTM step: gates = g @ Wih^T + h @ Whh^T + (bih + bhh), then pointwise
    cvt_f32_bf16_k<<<blk4(NN * HH), 256, 0, stream>>>(gbuf, gbufb, NN * HH / 4, 1.0f);
    gemm_k<true><<<gemm_grid_G, 32, 0, stream>>>(gbufb, hstateb, Wihb, Whhb, blstm, gates, HH, 4 * HH, 0);
    lstm_update_k<<<NN * HH / 256, 256, 0, stream>>>(gates, hstate, cstate);
    cvt_f32_bf16_k<<<blk4(NN * HH), 256, 0, stream>>>(hstate, hstateb, NN * HH / 4, 1.0f);
  }

  // predictor: out[N,F] = h_last @ W_pred^T + b_pred
  gemm_k<false><<<gemm_grid_P, 32, 0, stream>>>(hstateb, nullptr, Wpb, nullptr, bp, (float*)d_out, HH, FF, 0);
}
